// BackgroundForegroundNeRF_48120813584957
// MI455X (gfx1250) — matrix operand order repaired and checked by reading, not compiled, not run on hardware
//
#include <hip/hip_runtime.h>
#include <hip/hip_bf16.h>
#include <math.h>

typedef __attribute__((ext_vector_type(16))) _Float16 v16h;
typedef __attribute__((ext_vector_type(8)))  _Float16 v8h;
typedef __attribute__((ext_vector_type(8)))  float    v8f;

#define NW    4            // waves per block (wave32)
#define BLOCK (NW * 32)

// merged per-wave activation buffer: XT = [0,1536) stride 96, VG = [1536,2560) stride 64
#define ACT_VG 1536
#define ACT_ELEMS 2560

// f16 weight image offsets in LDS (element counts), all K-padded / row-padded
#define O_BGS0 0           // 64x64  (src 64x63, col63 = 0)
#define O_BGS1 4096        // 64x64
#define O_BGS2 8192        // 32x64  (src 17x64, rows 17..31 = 0)
#define O_BGC0 10240       // 64x64  (src 64x42, cols 42..63 = 0)
#define O_BGC1 14336       // 64x64
#define O_BGC2 18432       // 64x64
#define O_BGC3 22528       // 16x64  (src 3x64, rows 3..15 = 0)
#define O_FGS0 23552       // 64x96  (src 64x71, cols 71..95 = 0)
#define O_FGS1 29696       // 64x64
#define O_FGS2 33792       // 32x64
#define O_FGC0 35840       // 64x64
#define O_FGC1 39936       // 64x64
#define O_FGC2 44032       // 64x64
#define O_FGC3 48128       // 16x64
#define W_ELEMS 49152      // 96 KB of f16

__device__ inline v8f wmma_f16(v16h a, v16h b, v8f c) {
  return __builtin_amdgcn_wmma_f32_16x16x32_f16(
      /*neg_a=*/false, a, /*neg_b=*/false, b,
      /*c_mod=*/(short)0, c, /*reuse_a=*/false, /*reuse_b=*/false);
}

__device__ inline v16h cat8(v8h lo, v8h hi) {
  v16h r;
#pragma unroll
  for (int i = 0; i < 8; ++i) { r[i] = lo[i]; r[i + 8] = hi[i]; }
  return r;
}

// A fragment (16x32 f16): lane<16 holds row=lane, K = ks+{0..7} and ks+{16..23};
// lane>=16 holds row=lane-16, K = ks+{8..15} and ks+{24..31}.
__device__ inline v16h load_a(const _Float16* src, int stride, int lane, int ks) {
  const int r   = lane & 15;
  const int off = (lane < 16) ? 0 : 8;
  const _Float16* p = src + r * stride + ks + off;
  v8h lo = *(const v8h*)(p);
  v8h hi = *(const v8h*)(p + 16);
  return cat8(lo, hi);
}

// B fragment (32x16 f16) from row-major W [rows x K]: B[k][n] = W[nbase+n][k].
__device__ inline v16h load_b(const _Float16* w, int K, int lane, int nbase, int ks) {
  const int n   = lane & 15;
  const int off = (lane < 16) ? 0 : 8;
  const _Float16* p = w + (nbase + n) * K + ks + off;
  v8h lo = *(const v8h*)(p);
  v8h hi = *(const v8h*)(p + 16);
  return cat8(lo, hi);
}

// Hidden layer: dst[16x64] (stride 64) = relu(src[16xK] @ W^T), K in {64,96}.
__device__ inline void layer_h(const _Float16* src, int sstride, int K,
                               const _Float16* w, _Float16* dst, int lane) {
  const int nk = K >> 5;
  v16h a[3];
#pragma unroll
  for (int t = 0; t < 3; ++t)
    if (t < nk) a[t] = load_a(src, sstride, lane, t * 32);
  const int n     = lane & 15;
  const int rbase = (lane < 16) ? 0 : 8;
#pragma unroll
  for (int b = 0; b < 4; ++b) {
    v8f acc = {};
    for (int t = 0; t < nk; ++t)
      acc = wmma_f16(a[t], load_b(w, K, lane, b * 16, t * 32), acc);
#pragma unroll
    for (int v = 0; v < 8; ++v) {
      float hv = acc[v];
      hv = hv > 0.f ? hv : 0.f;
      dst[(rbase + v) * 64 + b * 16 + n] = (_Float16)hv;
    }
  }
}

// Sigma head: sd[16x32] f32 = src[16x64] @ W^T (W padded to 32 rows), no relu.
__device__ inline void layer_sd(const _Float16* src, const _Float16* w,
                                float* sdw, int lane) {
  v16h a0 = load_a(src, 64, lane, 0);
  v16h a1 = load_a(src, 64, lane, 32);
  const int n     = lane & 15;
  const int rbase = (lane < 16) ? 0 : 8;
#pragma unroll
  for (int b = 0; b < 2; ++b) {
    v8f acc = {};
    acc = wmma_f16(a0, load_b(w, 64, lane, b * 16, 0),  acc);
    acc = wmma_f16(a1, load_b(w, 64, lane, b * 16, 32), acc);
#pragma unroll
    for (int v = 0; v < 8; ++v) sdw[(rbase + v) * 32 + b * 16 + n] = acc[v];
  }
}

// Color head: sd[16x32] cols 0..15 = src[16x64] @ W^T (W padded to 16 rows).
__device__ inline void layer_c3(const _Float16* src, const _Float16* w,
                                float* sdw, int lane) {
  v16h a0 = load_a(src, 64, lane, 0);
  v16h a1 = load_a(src, 64, lane, 32);
  v8f acc = {};
  acc = wmma_f16(a0, load_b(w, 64, lane, 0, 0),  acc);
  acc = wmma_f16(a1, load_b(w, 64, lane, 0, 32), acc);
  const int n     = lane & 15;
  const int rbase = (lane < 16) ? 0 : 8;
#pragma unroll
  for (int v = 0; v < 8; ++v) sdw[(rbase + v) * 32 + n] = acc[v];
}

__device__ inline float softplus(float v) {
  return fmaxf(v, 0.f) + log1pf(expf(-fabsf(v)));
}

__device__ inline void cvt_w(_Float16* dst, const float* src, int rows, int K,
                             int srows, int scols, int tid) {
  for (int idx = tid; idx < rows * K; idx += BLOCK) {
    int r = idx / K, c = idx - r * K;
    float v = (r < srows && c < scols) ? src[r * scols + c] : 0.f;
    dst[idx] = (_Float16)v;
  }
}

__global__ __launch_bounds__(BLOCK) void nerf_bgfg_wmma(
    const float* __restrict__ x,
    const float* __restrict__ bgs0, const float* __restrict__ bgs1,
    const float* __restrict__ bgs2, const float* __restrict__ bgc0,
    const float* __restrict__ bgc1, const float* __restrict__ bgc2,
    const float* __restrict__ bgc3, const float* __restrict__ fgs0,
    const float* __restrict__ fgs1, const float* __restrict__ fgs2,
    const float* __restrict__ fgc0, const float* __restrict__ fgc1,
    const float* __restrict__ fgc2, const float* __restrict__ fgc3,
    float* __restrict__ out, int ntiles) {
  __shared__ __attribute__((aligned(16))) _Float16 wl[W_ELEMS];         // 96 KB weights
  __shared__ __attribute__((aligned(16))) _Float16 act[NW][ACT_ELEMS];  // pts | views+geo
  __shared__ __attribute__((aligned(16))) _Float16 ht[NW][16 * 64];     // hidden acts
  __shared__ __attribute__((aligned(16))) float    sd[NW][16 * 32];     // f32 head outputs

  const int tid = threadIdx.x;
  cvt_w(wl + O_BGS0, bgs0, 64, 64, 64, 63, tid);
  cvt_w(wl + O_BGS1, bgs1, 64, 64, 64, 64, tid);
  cvt_w(wl + O_BGS2, bgs2, 32, 64, 17, 64, tid);
  cvt_w(wl + O_BGC0, bgc0, 64, 64, 64, 42, tid);
  cvt_w(wl + O_BGC1, bgc1, 64, 64, 64, 64, tid);
  cvt_w(wl + O_BGC2, bgc2, 64, 64, 64, 64, tid);
  cvt_w(wl + O_BGC3, bgc3, 16, 64,  3, 64, tid);
  cvt_w(wl + O_FGS0, fgs0, 64, 96, 64, 71, tid);
  cvt_w(wl + O_FGS1, fgs1, 64, 64, 64, 64, tid);
  cvt_w(wl + O_FGS2, fgs2, 32, 64, 17, 64, tid);
  cvt_w(wl + O_FGC0, fgc0, 64, 64, 64, 42, tid);
  cvt_w(wl + O_FGC1, fgc1, 64, 64, 64, 64, tid);
  cvt_w(wl + O_FGC2, fgc2, 64, 64, 64, 64, tid);
  cvt_w(wl + O_FGC3, fgc3, 16, 64,  3, 64, tid);
  __syncthreads();

  const int wave = tid >> 5;
  const int lane = tid & 31;
  _Float16* ACT = act[wave];
  _Float16* HT  = ht[wave];
  float*    SD  = sd[wave];

  // one-time zero of padded activation columns (read only against zero weights,
  // but must be finite): XT cols 71..95, VG cols 27..63.
  for (int idx = lane; idx < 16 * 25; idx += 32) {
    int r = idx / 25, c = idx - r * 25;
    ACT[r * 96 + 71 + c] = (_Float16)0.f;
  }
  for (int idx = lane; idx < 16 * 37; idx += 32) {
    int r = idx / 37, c = idx - r * 37;
    ACT[ACT_VG + r * 64 + 27 + c] = (_Float16)0.f;
  }
  __builtin_amdgcn_wave_barrier();

  // per-point scalars live in registers of lane p (p = lane < 16)
  float r_bgs = 0.f, r_bc0 = 0.f, r_bc1 = 0.f, r_bc2 = 0.f;
  float r_fgs = 0.f, r_fgu = 0.f;

  for (int tile = blockIdx.x * NW + wave; tile < ntiles; tile += gridDim.x * NW) {
    const int p0 = tile << 4;
    // stage x tile: 1568 contiguous floats = 392 aligned float4 loads, branchless
    // routing: col<71 -> XT[r*96+c], else views -> VG[r*64 + (c-71)].
    {
      const float4* xr = (const float4*)(x + (size_t)p0 * 98);
      for (int q = lane; q < 392; q += 32) {
        float4 v = xr[q];
#pragma unroll
        for (int j = 0; j < 4; ++j) {
          unsigned ee = (unsigned)(q * 4 + j);
          unsigned r = ee / 98u;
          unsigned c = ee - r * 98u;
          int addr = (c < 71u) ? (int)(r * 96u + c)
                               : (int)(ACT_VG + r * 64u + (c - 71u));
          ACT[addr] = (_Float16)(((const float*)&v)[j]);
        }
      }
    }
    __builtin_amdgcn_wave_barrier();

    // ---- background: sigma net (xyz = pts[:,:63]; weight col63 zeroed) ----
    layer_h(ACT, 96, 64, wl + O_BGS0, HT, lane);
    layer_h(HT, 64, 64, wl + O_BGS1, HT, lane);
    layer_sd(HT, wl + O_BGS2, SD, lane);
    if (lane < 16) {
      r_bgs = softplus(SD[lane * 32 + 0]);                 // bg_sigma
#pragma unroll
      for (int j = 0; j < 15; ++j)                         // bg geo -> color input
        ACT[ACT_VG + lane * 64 + 27 + j] = (_Float16)SD[lane * 32 + 2 + j];
    }
    __builtin_amdgcn_wave_barrier();
    // ---- background: color net ----
    layer_h(ACT + ACT_VG, 64, 64, wl + O_BGC0, HT, lane);
    layer_h(HT, 64, 64, wl + O_BGC1, HT, lane);
    layer_h(HT, 64, 64, wl + O_BGC2, HT, lane);
    layer_c3(HT, wl + O_BGC3, SD, lane);
    if (lane < 16) {
      r_bc0 = SD[lane * 32 + 0];
      r_bc1 = SD[lane * 32 + 1];
      r_bc2 = SD[lane * 32 + 2];
    }
    __builtin_amdgcn_wave_barrier();

    // ---- foreground: sigma net (full pts, K padded 71->96) ----
    layer_h(ACT, 96, 96, wl + O_FGS0, HT, lane);
    layer_h(HT, 64, 64, wl + O_FGS1, HT, lane);
    layer_sd(HT, wl + O_FGS2, SD, lane);
    if (lane < 16) {
      r_fgs = softplus(SD[lane * 32 + 0]);                 // fg_sigma
      r_fgu = softplus(SD[lane * 32 + 1]);                 // fg_unc
#pragma unroll
      for (int j = 0; j < 15; ++j)                         // fg geo overwrites bg geo
        ACT[ACT_VG + lane * 64 + 27 + j] = (_Float16)SD[lane * 32 + 2 + j];
    }
    __builtin_amdgcn_wave_barrier();
    // ---- foreground: color net ----
    layer_h(ACT + ACT_VG, 64, 64, wl + O_FGC0, HT, lane);
    layer_h(HT, 64, 64, wl + O_FGC1, HT, lane);
    layer_h(HT, 64, 64, wl + O_FGC2, HT, lane);
    layer_c3(HT, wl + O_FGC3, SD, lane);

    // ---- blend + write 6 f32 per point (3 x float2, 8B aligned) ----
    if (lane < 16) {
      const int p = lane;
      float sig = r_bgs + r_fgs + 1e-9f;
      float wb = r_bgs / sig, wf = r_fgs / sig;
      float2* op = (float2*)(out + (size_t)(p0 + p) * 6);
      float2 o0, o1, o2;
      o0.x = wb * r_bc0 + wf * SD[p * 32 + 0];
      o0.y = wb * r_bc1 + wf * SD[p * 32 + 1];
      o1.x = wb * r_bc2 + wf * SD[p * 32 + 2];
      o1.y = sig;
      o2.x = r_fgu;
      o2.y = r_fgs;
      op[0] = o0; op[1] = o1; op[2] = o2;
    }
    __builtin_amdgcn_wave_barrier();
  }
}

extern "C" void kernel_launch(void* const* d_in, const int* in_sizes, int n_in,
                              void* d_out, int out_size, void* d_ws, size_t ws_size,
                              hipStream_t stream) {
  const float* x    = (const float*)d_in[0];
  const float* bgs0 = (const float*)d_in[1];
  const float* bgs1 = (const float*)d_in[2];
  const float* bgs2 = (const float*)d_in[3];
  const float* bgc0 = (const float*)d_in[4];
  const float* bgc1 = (const float*)d_in[5];
  const float* bgc2 = (const float*)d_in[6];
  const float* bgc3 = (const float*)d_in[7];
  const float* fgs0 = (const float*)d_in[8];
  const float* fgs1 = (const float*)d_in[9];
  const float* fgs2 = (const float*)d_in[10];
  const float* fgc0 = (const float*)d_in[11];
  const float* fgc1 = (const float*)d_in[12];
  const float* fgc2 = (const float*)d_in[13];
  const float* fgc3 = (const float*)d_in[14];

  const int npts   = in_sizes[0] / 98;   // x is (N, 98)
  const int ntiles = npts / 16;
  int grid = 4096;
  int maxg = (ntiles + NW - 1) / NW;
  if (grid > maxg) grid = maxg;
  if (grid < 1) grid = 1;

  nerf_bgfg_wmma<<<grid, BLOCK, 0, stream>>>(
      x, bgs0, bgs1, bgs2, bgc0, bgc1, bgc2, bgc3,
      fgs0, fgs1, fgs2, fgc0, fgc1, fgc2, fgc3,
      (float*)d_out, ntiles);
}
